// mixprop_6682969112724
// MI455X (gfx1250) — matrix operand order repaired and checked by reading, not compiled, not run on hardware
//
#include <hip/hip_runtime.h>

typedef __attribute__((ext_vector_type(16))) _Float16 v16h;
typedef __attribute__((ext_vector_type(8)))  _Float16 h8;
typedef __attribute__((ext_vector_type(8)))  float    v8f;

#define MP_B    32
#define MP_N    512
#define MP_L    64
#define MP_C    32
#define MP_CO   32
#define MP_K    128          // (GDEP+1)*C
#define MP_ALPHA 0.05f

// ---------------------------------------------------------------------------
// Kernel 1: per-node coefficients.  coef[n][k] for k=0..3.
// rs[n] = sum_v (adj[n,v] + I)/d[n],  d[n] = sum_v adj[n,v] + 1
// coef_0 = 1 ; coef_k = alpha + (1-alpha)*rs*coef_{k-1}
// One wave32 per node.
// ---------------------------------------------------------------------------
__global__ __launch_bounds__(32) void prep_coef(const float* __restrict__ adj,
                                                float* __restrict__ coef) {
    const int n    = blockIdx.x;
    const int lane = threadIdx.x;

    float s = 0.0f;
    for (int v = lane; v < MP_N; v += 32) s += adj[n * MP_N + v];
    #pragma unroll
    for (int off = 16; off > 0; off >>= 1) s += __shfl_xor(s, off, 32);
    const float d = s + 1.0f;                      // self loop

    float t = 0.0f;
    for (int v = lane; v < MP_N; v += 32) {
        float a = adj[n * MP_N + v] + (v == n ? 1.0f : 0.0f);
        t += a / d;
    }
    #pragma unroll
    for (int off = 16; off > 0; off >>= 1) t += __shfl_xor(t, off, 32);

    if (lane == 0) {
        const float rs = t;
        float c = 1.0f;
        coef[n * 4 + 0] = c;
        #pragma unroll
        for (int k = 1; k < 4; ++k) {
            c = MP_ALPHA + (1.0f - MP_ALPHA) * rs * c;
            coef[n * 4 + k] = c;
        }
    }
}

// ---------------------------------------------------------------------------
// Kernel 2: W fp32 -> f16  (32 x 128 = 4096 elements)
// ---------------------------------------------------------------------------
__global__ __launch_bounds__(256) void prep_w(const float* __restrict__ W,
                                              _Float16* __restrict__ Wh) {
    int i = blockIdx.x * 256 + threadIdx.x;
    if (i < MP_CO * MP_K) Wh[i] = (_Float16)W[i];
}

// ---------------------------------------------------------------------------
// Kernel 3: main GEMM.  One 256-thread block (8 waves) per (b,l) slab.
//   Out[32 x 512] = W[32 x 128] x Bs[128 x 512] + bias
//   Bs[k*32+c, n] = coef_k(n) * x[b,n,l,c]   (f16, scaled at fragment build)
// x slab staged to LDS with CDNA5 async global->LDS DMA (ASYNCcnt path);
// per n-tile: 2 M-tiles x 4 K-steps of v_wmma_f32_16x16x32_f16.
// ---------------------------------------------------------------------------
__global__ __launch_bounds__(256) void mixprop_main(
        const float*    __restrict__ x,
        const float*    __restrict__ coef,
        const _Float16* __restrict__ Wh,
        const float*    __restrict__ bias,
        float*          __restrict__ out) {
    const int l   = blockIdx.x;          // 0..63
    const int b   = blockIdx.y;          // 0..31
    const int tid = threadIdx.x;

    __shared__ __align__(16) float    xshf[MP_N * MP_C];  // 64 KB, [n][c] f32
    __shared__ _Float16               csh[MP_N * 4];      //  4 KB, [n][k] f16

    // ---- async-stage x[b, :, l, :]: 512 rows x 128B, cache -> LDS DMA ----
    // global_load_async_to_lds_b128: vdst = LDS byte offset, vaddr = 64b addr.
    // Tracked on ASYNCcnt; no VGPR round-trip, no ds_store traffic.
    const float* xb = x + (size_t)b * MP_N * MP_L * MP_C + (size_t)l * MP_C;
    #pragma unroll
    for (int i = 0; i < 16; ++i) {
        const int q = i * 256 + tid;        // 0..4095 16B chunks
        const int n = q >> 3;
        const int j = q & 7;
        const float* gp = xb + (size_t)n * (MP_L * MP_C) + j * 4;
        // low 32 bits of a generic LDS pointer == LDS byte offset
        unsigned laddr = (unsigned)(uintptr_t)(&xshf[n * MP_C + j * 4]);
        asm volatile("global_load_async_to_lds_b128 %0, %1, off"
                     :: "v"(laddr), "v"(gp) : "memory");
    }
    // ---- stage coef as f16 (tiny, normal path) ----
    #pragma unroll
    for (int i = 0; i < 8; ++i) {
        const int q = i * 256 + tid;        // 0..2047
        csh[q] = (_Float16)coef[q];
    }
    asm volatile("s_wait_asynccnt 0x0" ::: "memory");
    __syncthreads();

    const int wave  = tid >> 5;
    const int lane  = tid & 31;
    const int lhalf = lane >> 4;            // 0 | 1
    const int l15   = lane & 15;

    // ---- A fragments: W in contiguous k order (K = 16*lhalf + s) ----
    // lane<16 : M=l15, halves 0..15 = K 0..15
    // lane>=16: M=l15, halves 0..15 = K 16..31
    v16h afrag[2][4];
    #pragma unroll
    for (int m = 0; m < 2; ++m) {
        #pragma unroll
        for (int j = 0; j < 4; ++j) {
            const int o = m * 16 + l15;
            const _Float16* p = Wh + o * MP_K + j * 32 + lhalf * 16;
            const h8 lo = *(const h8*)(p);
            const h8 hi = *(const h8*)(p + 8);
            v16h a;
            #pragma unroll
            for (int e = 0; e < 8; ++e) { a[e] = lo[e]; a[8 + e] = hi[e]; }
            afrag[m][j] = a;
        }
    }

    // ---- bias per D-matrix layout: VGPR r -> M = (lane<16 ? r : 8+r) ----
    float bias0[8], bias1[8];
    #pragma unroll
    for (int r = 0; r < 8; ++r) {
        bias0[r] = bias[lhalf * 8 + r];
        bias1[r] = bias[16 + lhalf * 8 + r];
    }

    // ---- 4 n-tiles per wave ----
    #pragma unroll
    for (int nt = 0; nt < 4; ++nt) {
        const int ntile = wave * 4 + nt;
        const int n     = ntile * 16 + l15;

        // B fragment base (unscaled x): contiguous k order, as A:
        // lane<16: N=l15, halves = K 0..15 ; lane>=16: N=l15, halves = K 16..31
        const float* xpf = &xshf[n * MP_C + lhalf * 16];
        const float4 f0 = *(const float4*)(xpf);
        const float4 f1 = *(const float4*)(xpf + 4);
        const float4 f2 = *(const float4*)(xpf + 8);
        const float4 f3 = *(const float4*)(xpf + 12);
        v16h xv;
        xv[0]  = (_Float16)f0.x; xv[1]  = (_Float16)f0.y;
        xv[2]  = (_Float16)f0.z; xv[3]  = (_Float16)f0.w;
        xv[4]  = (_Float16)f1.x; xv[5]  = (_Float16)f1.y;
        xv[6]  = (_Float16)f1.z; xv[7]  = (_Float16)f1.w;
        xv[8]  = (_Float16)f2.x; xv[9]  = (_Float16)f2.y;
        xv[10] = (_Float16)f2.z; xv[11] = (_Float16)f2.w;
        xv[12] = (_Float16)f3.x; xv[13] = (_Float16)f3.y;
        xv[14] = (_Float16)f3.z; xv[15] = (_Float16)f3.w;

        _Float16 ch[4];
        #pragma unroll
        for (int j = 0; j < 4; ++j) ch[j] = csh[n * 4 + j];

        v8f acc0 = {0.f, 0.f, 0.f, 0.f, 0.f, 0.f, 0.f, 0.f};
        v8f acc1 = {0.f, 0.f, 0.f, 0.f, 0.f, 0.f, 0.f, 0.f};

        #pragma unroll
        for (int j = 0; j < 4; ++j) {
            v16h bs;                         // coef_j(n) * x  (v_pk_mul_f16)
            #pragma unroll
            for (int e = 0; e < 16; ++e) bs[e] = xv[e] * ch[j];
            acc0 = __builtin_amdgcn_wmma_f32_16x16x32_f16(
                       false, afrag[0][j], false, bs, (short)0, acc0, false, false);
            acc1 = __builtin_amdgcn_wmma_f32_16x16x32_f16(
                       false, afrag[1][j], false, bs, (short)0, acc1, false, false);
        }

        // ---- epilogue: bias + store out[b,o,l,n] ----
        #pragma unroll
        for (int r = 0; r < 8; ++r) {
            const int o0 = lhalf * 8 + r;          // M-tile 0
            const int o1 = 16 + lhalf * 8 + r;     // M-tile 1
            out[((size_t)(b * MP_CO + o0) * MP_L + l) * MP_N + n] = acc0[r] + bias0[r];
            out[((size_t)(b * MP_CO + o1) * MP_L + l) * MP_N + n] = acc1[r] + bias1[r];
        }
    }
}

// ---------------------------------------------------------------------------
extern "C" void kernel_launch(void* const* d_in, const int* in_sizes, int n_in,
                              void* d_out, int out_size, void* d_ws, size_t ws_size,
                              hipStream_t stream) {
    const float* x    = (const float*)d_in[0];
    const float* adj  = (const float*)d_in[1];
    const float* W    = (const float*)d_in[2];
    const float* bias = (const float*)d_in[3];
    float*       out  = (float*)d_out;

    float*    coef = (float*)d_ws;                                   // 512*4 f32
    _Float16* Wh   = (_Float16*)((char*)d_ws + MP_N * 4 * sizeof(float)); // 4096 f16

    prep_coef<<<MP_N, 32, 0, stream>>>(adj, coef);
    prep_w<<<(MP_CO * MP_K + 255) / 256, 256, 0, stream>>>(W, Wh);
    mixprop_main<<<dim3(MP_L, MP_B), 256, 0, stream>>>(x, coef, Wh, bias, out);
}
